// StepAttention_33724083208694
// MI455X (gfx1250) — matrix operand order repaired and checked by reading, not compiled, not run on hardware
//
#include <hip/hip_runtime.h>
#include <hip/hip_bf16.h>
#include <math.h>

// ---------------------------------------------------------------------------
// Problem: B=8, T=4096, D=1024  (fp32 in / fp32 out)
//   texts_K = tanh(value @ W_k^T + b_k); scores = texts_K @ w_q
//   e = exp(scores - rowmax); out = cumsum(e*value)/cumsum(e)
// ---------------------------------------------------------------------------

#define BT   32768   // B*T
#define DIM  1024
#define TT   4096
#define NB   8
#define MT_ROWS 64   // M-tile rows per block in the WMMA kernel
#define NCH  32      // chunks per batch for the scan
#define TC   128     // timesteps per chunk

typedef __attribute__((ext_vector_type(16))) _Float16 v16h;
typedef __attribute__((ext_vector_type(8)))  _Float16 v8h;
typedef __attribute__((ext_vector_type(8)))  float    v8f;

// CDNA5 has a hardware V_TANH_F32 transcendental; use it when exposed.
static __device__ __forceinline__ float fast_tanh(float x) {
#if __has_builtin(__builtin_amdgcn_tanhf)
    return __builtin_amdgcn_tanhf(x);
#else
    // branch-free fallback: tanh(x) = (e-1)/(e+1), e = exp2(2*log2(e)*x)
    float e = __builtin_amdgcn_exp2f(2.885390082f * x); // 2/ln(2)
    return (e - 1.0f) * __builtin_amdgcn_rcpf(e + 1.0f);
#endif
}

// ===========================================================================
// Kernel 1: scores[m] = sum_e w_q[e] * tanh( value[m,:] . W_k[e,:] + b_k[e] )
// WMMA f32_16x16x32_f16, f32 accumulate.  grid = BT/64, block = 256 (8 waves)
// ===========================================================================
__global__ __launch_bounds__(256) void k_scores_wmma(
    const float* __restrict__ value,
    const float* __restrict__ Wk,
    const float* __restrict__ bk,
    const float* __restrict__ wq,
    float* __restrict__ scores)
{
    __shared__ _Float16 ldsA[MT_ROWS * DIM];   // 128 KB f16 A-tile
    __shared__ float    ldsScore[8 * MT_ROWS]; // per-wave partial rows

    const int tid  = threadIdx.x;
    const int wave = tid >> 5;
    const int lane = tid & 31;
    const long m0  = (long)blockIdx.x * MT_ROWS;

    // ---- stage value[m0 .. m0+63, :] into LDS as f16 (coalesced float4) ----
    {
        const float4* src = (const float4*)(value + m0 * DIM);
        #pragma unroll 4
        for (int i = 0; i < (MT_ROWS * DIM / 4) / 256; ++i) {
            int f4 = i * 256 + tid;
            float4 v = src[f4];
            _Float16* dst = &ldsA[f4 * 4];
            dst[0] = (_Float16)v.x; dst[1] = (_Float16)v.y;
            dst[2] = (_Float16)v.z; dst[3] = (_Float16)v.w;
        }
    }
    __syncthreads();

    const int n    = lane & 15;   // N column within 16x16 tile / B row id
    const int half = lane >> 4;   // lane half selects K range

    float partial[4][8];
    #pragma unroll
    for (int mt = 0; mt < 4; ++mt)
        #pragma unroll
        for (int r = 0; r < 8; ++r) partial[mt][r] = 0.f;

    // each wave covers E-tiles: (et*8 + wave), et = 0..7  -> 64 tiles total
    for (int et = 0; et < 8; ++et) {
        const int e0   = (et * 8 + wave) * 16;
        const float bkv = bk[e0 + n];
        const float wqv = wq[e0 + n];
        // B fragment source row: W_k[e0+n, :], this lane-half covers K+16*half
        const float* wrow = Wk + (long)(e0 + n) * DIM + half * 16;

        v8f acc[4] = {};   // 4 M-subtiles of 16x16 f32

        for (int k0 = 0; k0 < DIM; k0 += 32) {
            // hint next K-block of this W_k row into cache
            __builtin_prefetch(wrow + k0 + 32, 0, 1);

            // ---- B fragment: 32(K)x16(N) f16; lane reads 16 contiguous K ----
            const float4* wp = (const float4*)(wrow + k0);
            float4 w0 = wp[0], w1 = wp[1], w2 = wp[2], w3 = wp[3];
            v16h bf;
            bf[0]=(_Float16)w0.x;  bf[1]=(_Float16)w0.y;  bf[2]=(_Float16)w0.z;  bf[3]=(_Float16)w0.w;
            bf[4]=(_Float16)w1.x;  bf[5]=(_Float16)w1.y;  bf[6]=(_Float16)w1.z;  bf[7]=(_Float16)w1.w;
            bf[8]=(_Float16)w2.x;  bf[9]=(_Float16)w2.y;  bf[10]=(_Float16)w2.z; bf[11]=(_Float16)w2.w;
            bf[12]=(_Float16)w3.x; bf[13]=(_Float16)w3.y; bf[14]=(_Float16)w3.z; bf[15]=(_Float16)w3.w;

            const int ka = k0 + half * 16;  // A lane-half K base (contiguous, matches B)
            #pragma unroll
            for (int mt = 0; mt < 4; ++mt) {
                // ---- A fragment from LDS: 16(M)x32(K) f16 ----
                const _Float16* arow = &ldsA[(mt * 16 + n) * DIM];
                union { v16h v; v8h h[2]; } af;
                af.h[0] = *(const v8h*)(arow + ka);       // K = ka .. ka+7
                af.h[1] = *(const v8h*)(arow + ka + 8);   // K = ka+8 .. ka+15
                acc[mt] = __builtin_amdgcn_wmma_f32_16x16x32_f16(
                    false, af.v, false, bf, (short)0, acc[mt], false, false);
            }
        }

        // epilogue for this E-tile: tanh + w_q weighting, fold into partials
        #pragma unroll
        for (int mt = 0; mt < 4; ++mt)
            #pragma unroll
            for (int r = 0; r < 8; ++r)
                partial[mt][r] += wqv * fast_tanh(acc[mt][r] + bkv);
    }

    // reduce over the 16 N-columns held by each lane-half (xor shuffles)
    #pragma unroll
    for (int mt = 0; mt < 4; ++mt) {
        #pragma unroll
        for (int r = 0; r < 8; ++r) {
            float v = partial[mt][r];
            v += __shfl_xor(v, 1, 32);
            v += __shfl_xor(v, 2, 32);
            v += __shfl_xor(v, 4, 32);
            v += __shfl_xor(v, 8, 32);
            partial[mt][r] = v;
        }
    }
    // lanes 0 and 16 hold the sums for rows r and r+8 of each sub-tile.
    // deterministic cross-wave combine via LDS (no FP atomics).
    if (n == 0) {
        #pragma unroll
        for (int mt = 0; mt < 4; ++mt)
            #pragma unroll
            for (int r = 0; r < 8; ++r)
                ldsScore[wave * MT_ROWS + mt * 16 + half * 8 + r] = partial[mt][r];
    }
    __syncthreads();
    if (tid < MT_ROWS) {
        float s = 0.f;
        #pragma unroll
        for (int w = 0; w < 8; ++w) s += ldsScore[w * MT_ROWS + tid];
        scores[m0 + tid] = s;
    }
}

// ===========================================================================
// Kernel 2: per batch: m = max(scores); e = exp(s-m); den = cumsum(e)
// grid = B, block = 256, each thread owns 16 contiguous timesteps
// ===========================================================================
__global__ __launch_bounds__(256) void k_softmax_prep(
    const float* __restrict__ scores,
    float* __restrict__ e,
    float* __restrict__ den)
{
    __shared__ float red[256];
    const int b = blockIdx.x, tid = threadIdx.x;
    const float* s = scores + b * TT;

    float mx = -INFINITY;
    #pragma unroll
    for (int i = 0; i < 16; ++i) mx = fmaxf(mx, s[tid * 16 + i]);
    red[tid] = mx; __syncthreads();
    for (int o = 128; o > 0; o >>= 1) {
        if (tid < o) red[tid] = fmaxf(red[tid], red[tid + o]);
        __syncthreads();
    }
    const float m = red[0];
    __syncthreads();

    // args are always <= 0, so the fast hardware exp path is safe
    float th = 0.f;
    #pragma unroll
    for (int i = 0; i < 16; ++i) th += __expf(s[tid * 16 + i] - m);
    red[tid] = th; __syncthreads();
    // Hillis-Steele inclusive scan of the 256 per-thread sums
    for (int o = 1; o < 256; o <<= 1) {
        float add = (tid >= o) ? red[tid - o] : 0.f;
        __syncthreads();
        red[tid] += add;
        __syncthreads();
    }
    float run = (tid > 0) ? red[tid - 1] : 0.f;  // exclusive prefix
    #pragma unroll
    for (int i = 0; i < 16; ++i) {
        int t = tid * 16 + i;
        float ev = __expf(s[t] - m);
        e[b * TT + t] = ev;
        run += ev;
        den[b * TT + t] = run;
    }
}

// ===========================================================================
// Kernel 3: chunk sums  csum[b,c,d] = sum_{t in chunk} e[t]*value[b,t,d]
// grid = (NCH, B), block = 256; thread owns d, d+256, d+512, d+768
// ===========================================================================
__global__ __launch_bounds__(256) void k_chunk_sums(
    const float* __restrict__ value,
    const float* __restrict__ e,
    float* __restrict__ csum)
{
    __shared__ float eLds[TC];
    const int chunk = blockIdx.x, b = blockIdx.y, tid = threadIdx.x;
    const int t0 = chunk * TC;
    if (tid < TC) eLds[tid] = e[b * TT + t0 + tid];
    __syncthreads();

    const float* vp = value + ((long)b * TT + t0) * DIM;
    float s0 = 0.f, s1 = 0.f, s2 = 0.f, s3 = 0.f;
    for (int t = 0; t < TC; ++t) {
        float ev = eLds[t];
        const float* row = vp + (long)t * DIM;
        s0 += ev * row[tid];
        s1 += ev * row[tid + 256];
        s2 += ev * row[tid + 512];
        s3 += ev * row[tid + 768];
    }
    float* cp = csum + (long)(b * NCH + chunk) * DIM;
    cp[tid] = s0; cp[tid + 256] = s1; cp[tid + 512] = s2; cp[tid + 768] = s3;
}

// ===========================================================================
// Kernel 4: in-place exclusive scan of csum over the chunk dim, per (b,d)
// grid = B*DIM/256 = 32, block = 256
// ===========================================================================
__global__ __launch_bounds__(256) void k_chunk_scan(float* __restrict__ csum)
{
    const int id = blockIdx.x * 256 + threadIdx.x;   // 0 .. 8191
    const int b = id >> 10, d = id & (DIM - 1);
    float run = 0.f;
    for (int c = 0; c < NCH; ++c) {
        long idx = (long)(b * NCH + c) * DIM + d;
        float v = csum[idx];
        csum[idx] = run;
        run += v;
    }
}

// ===========================================================================
// Kernel 5: out[b,t,d] = (prefix[b,chunk,d] + local cumsum e*value) / den[b,t]
// grid = (NCH, B), block = 256
// ===========================================================================
__global__ __launch_bounds__(256) void k_finalize(
    const float* __restrict__ value,
    const float* __restrict__ e,
    const float* __restrict__ den,
    const float* __restrict__ csum,
    float* __restrict__ out)
{
    __shared__ float eLds[TC];
    __shared__ float dLds[TC];
    const int chunk = blockIdx.x, b = blockIdx.y, tid = threadIdx.x;
    const int t0 = chunk * TC;
    if (tid < TC) {
        eLds[tid] = e[b * TT + t0 + tid];
        dLds[tid] = den[b * TT + t0 + tid];
    }
    __syncthreads();

    const float* cp = csum + (long)(b * NCH + chunk) * DIM;
    float r0 = cp[tid], r1 = cp[tid + 256], r2 = cp[tid + 512], r3 = cp[tid + 768];

    const float* vp = value + ((long)b * TT + t0) * DIM;
    float*       op = out   + ((long)b * TT + t0) * DIM;
    for (int t = 0; t < TC; ++t) {
        float ev  = eLds[t];
        float inv = __builtin_amdgcn_rcpf(dLds[t]);   // v_rcp_f32 (den >= e_max > 0)
        const float* row  = vp + (long)t * DIM;
        float*       orow = op + (long)t * DIM;
        r0 += ev * row[tid];       orow[tid]       = r0 * inv;
        r1 += ev * row[tid + 256]; orow[tid + 256] = r1 * inv;
        r2 += ev * row[tid + 512]; orow[tid + 512] = r2 * inv;
        r3 += ev * row[tid + 768]; orow[tid + 768] = r3 * inv;
    }
}

// ===========================================================================
extern "C" void kernel_launch(void* const* d_in, const int* in_sizes, int n_in,
                              void* d_out, int out_size, void* d_ws, size_t ws_size,
                              hipStream_t stream)
{
    const float* value = (const float*)d_in[0];  // [B,T,D]
    const float* Wk    = (const float*)d_in[1];  // [D,D]
    const float* bk    = (const float*)d_in[2];  // [D]
    const float* wq    = (const float*)d_in[3];  // [D]
    float* out = (float*)d_out;                  // [B,T,D]

    // workspace layout (floats): scores | e | den | csum
    float* scores = (float*)d_ws;                    // BT
    float* e      = scores + BT;                     // BT
    float* den    = e + BT;                          // BT
    float* csum   = den + BT;                        // B*NCH*DIM

    k_scores_wmma <<<BT / MT_ROWS, 256, 0, stream>>>(value, Wk, bk, wq, scores);
    k_softmax_prep<<<NB, 256, 0, stream>>>(scores, e, den);
    k_chunk_sums  <<<dim3(NCH, NB), 256, 0, stream>>>(value, e, csum);
    k_chunk_scan  <<<(NB * DIM) / 256, 256, 0, stream>>>(csum);
    k_finalize    <<<dim3(NCH, NB), 256, 0, stream>>>(value, e, den, csum, out);
}
